// Decoder_40338332844507
// MI455X (gfx1250) — matrix operand order repaired and checked by reading, not compiled, not run on hardware
//
#include <hip/hip_runtime.h>

// CDNA5 / gfx1250: wave32, WMMA 16x16x32 bf16, f32 accumulate.
// bf16x3 split GEMM: out[b,p,t,w] = sum_c q[b,t,c]*W[w,p,c] + bias[w,p]
// Pass 1: split W -> bf16 hi/lo in workspace (once).
// Pass 2: per block, split a 64-row q tile into LDS once, hoist A fragments
//         into VGPRs, then sweep all 49 N-pairs. B fragments are fed from
//         global (L0/L2 hot) through a depth-2 software pipeline so WMMAs
//         never wait on loadcnt==0.

typedef __attribute__((ext_vector_type(16))) __bf16 v16bf;
typedef __attribute__((ext_vector_type(8)))  __bf16 v8bf;
typedef __attribute__((ext_vector_type(8)))  float  v8f;

#define T_DIM   2048
#define C_DIM   256
#define P_DIM   32
#define W_DIM   49
#define NFLAT   1568                  // W_DIM * P_DIM columns
#define WELEMS  (NFLAT * C_DIM)       // 401408 weight elements
#define M_TILE  64
#define AP      264                   // padded LDS row stride (bf16 elems)

__device__ __forceinline__ unsigned short f32_to_bf16_rne(float f) {
    unsigned u = __builtin_bit_cast(unsigned, f);
    unsigned r = u + 0x7FFFu + ((u >> 16) & 1u);   // round-to-nearest-even
    return (unsigned short)(r >> 16);
}
__device__ __forceinline__ float bf16_bits_to_f32(unsigned short h) {
    return __builtin_bit_cast(float, (unsigned)h << 16);
}
__device__ __forceinline__ void split_bf16(float f, unsigned short& hi, unsigned short& lo) {
    hi = f32_to_bf16_rne(f);
    lo = f32_to_bf16_rne(f - bf16_bits_to_f32(hi));
}

// Load one 16x32 B fragment pair (hi and lo planes) for this lane.
__device__ __forceinline__ void load_bfrag(v16bf& bh, v16bf& bl,
                                           const unsigned short* __restrict__ BrH,
                                           const unsigned short* __restrict__ BrL,
                                           int off) {
    v8bf b0h = *(const v8bf*)&BrH[off];
    v8bf b1h = *(const v8bf*)&BrH[off + 8];
    v8bf b0l = *(const v8bf*)&BrL[off];
    v8bf b1l = *(const v8bf*)&BrL[off + 8];
    bh = __builtin_shufflevector(b0h, b1h, 0,1,2,3,4,5,6,7,8,9,10,11,12,13,14,15);
    bl = __builtin_shufflevector(b0l, b1l, 0,1,2,3,4,5,6,7,8,9,10,11,12,13,14,15);
}

// ---- Pass 1: split W [49,32,256] f32 -> bf16 hi/lo (row order preserved) ----
__global__ __launch_bounds__(256) void split_weights(
    const float* __restrict__ Wt,
    unsigned short* __restrict__ WH,
    unsigned short* __restrict__ WL)
{
    const int i4 = blockIdx.x * 256 + threadIdx.x;   // float4 index, 100352 total
    float4 v = ((const float4*)Wt)[i4];
    ushort4 h, l;
    split_bf16(v.x, h.x, l.x);
    split_bf16(v.y, h.y, l.y);
    split_bf16(v.z, h.z, l.z);
    split_bf16(v.w, h.w, l.w);
    *(ushort4*)&WH[i4 * 4] = h;
    *(ushort4*)&WL[i4 * 4] = l;
}

// ---- Pass 2: the GEMM ----
__global__ __launch_bounds__(256) void decoder_head_gemm(
    const float* __restrict__ q,              // [65536, 256]
    const unsigned short* __restrict__ WH,    // [1568, 256] bf16 hi bits
    const unsigned short* __restrict__ WL,    // [1568, 256] bf16 lo bits
    const float* __restrict__ bias,           // [49, 32]
    float* __restrict__ out)                  // [32, 32, 2048, 49]
{
    __shared__ unsigned short Ah[M_TILE * AP];
    __shared__ unsigned short Al[M_TILE * AP];

    const int tid  = threadIdx.x;
    const int lane = tid & 31;
    const int wave = tid >> 5;
    const int wm   = wave & 3;    // 16-row subtile 0..3
    const int wn   = wave >> 2;   // 16-col subtile 0..1

    const int mBase = blockIdx.x * M_TILE;

    // ---- Split q tile 64x256 -> bf16 hi/lo in LDS (once per block) ----
    {
        const float4* q4 = (const float4*)q;
        #pragma unroll
        for (int j = 0; j < 16; ++j) {
            int e4  = j * 256 + tid;
            int el  = e4 * 4;
            int r   = el >> 8;
            int col = el & 255;
            float4 v = q4[(size_t)(mBase + r) * (C_DIM / 4) + (col >> 2)];
            ushort4 h, l;
            split_bf16(v.x, h.x, l.x);
            split_bf16(v.y, h.y, l.y);
            split_bf16(v.z, h.z, l.z);
            split_bf16(v.w, h.w, l.w);
            *(ushort4*)&Ah[r * AP + col] = h;
            *(ushort4*)&Al[r * AP + col] = l;
        }
    }
    __syncthreads();

    // Fragment addressing (contiguous k order, this kernel's choice): A lane L -> row L&15, K base +16 for L>=16;
    // B lane L -> col L&15, K base +16 for L>=16.
    const int r16  = lane & 15;
    const int hi16 = lane >> 4;
    const int aoff = hi16 * 16;
    const int boff = hi16 * 16;

    // ---- Hoist all A fragments (8 K-chunks, hi+lo) into VGPRs ----
    const unsigned short* ArowH = &Ah[(wm * 16 + r16) * AP];
    const unsigned short* ArowL = &Al[(wm * 16 + r16) * AP];
    v16bf ahf[8], alf[8];
    #pragma unroll
    for (int kc = 0; kc < 8; ++kc) {
        const int ka = kc * 32;
        v8bf a0h = *(const v8bf*)&ArowH[ka + aoff];
        v8bf a1h = *(const v8bf*)&ArowH[ka + 8 + aoff];
        v8bf a0l = *(const v8bf*)&ArowL[ka + aoff];
        v8bf a1l = *(const v8bf*)&ArowL[ka + 8 + aoff];
        ahf[kc] = __builtin_shufflevector(a0h, a1h, 0,1,2,3,4,5,6,7,8,9,10,11,12,13,14,15);
        alf[kc] = __builtin_shufflevector(a0l, a1l, 0,1,2,3,4,5,6,7,8,9,10,11,12,13,14,15);
    }

    // Per-lane output row base (constant across the N sweep)
    const int m0 = mBase + wm * 16 + hi16 * 8;
    const int bb = m0 >> 11;            // / T_DIM
    const int t0 = m0 & (T_DIM - 1);

    // ---- Depth-2 pipelined sweep over all 49 N-pairs ----
    const int n0 = wn * 16 + r16;                     // lane's column at it=0
    const unsigned short* BrH = WH + (size_t)n0 * C_DIM;
    const unsigned short* BrL = WL + (size_t)n0 * C_DIM;

    v16bf bhb[4], blb[4];                             // 4 rotating fragment buffers
    load_bfrag(bhb[0], blb[0], BrH, BrL, 0 * 32 + boff);   // chunk 0
    load_bfrag(bhb[1], blb[1], BrH, BrL, 1 * 32 + boff);   // chunk 1

    for (int it = 0; it < 49; ++it) {
        // Next iteration's row pointers (clamped at the end; redundant load, no OOB)
        const unsigned short* nBrH = (it < 48) ? BrH + 32 * C_DIM : BrH;
        const unsigned short* nBrL = (it < 48) ? BrL + 32 * C_DIM : BrL;

        v8f acc = {};
        #pragma unroll
        for (int kc = 0; kc < 8; ++kc) {
            const int cur = kc & 3;
            const int pre = (kc + 2) & 3;
            // Issue chunk kc+2 (wrapping into the next N-pair) before computing
            if (kc < 6) {
                load_bfrag(bhb[pre], blb[pre], BrH, BrL, (kc + 2) * 32 + boff);
            } else {
                load_bfrag(bhb[pre], blb[pre], nBrH, nBrL, (kc - 6) * 32 + boff);
            }
            acc = __builtin_amdgcn_wmma_f32_16x16x32_bf16(false, ahf[kc], false, blb[cur], (short)0, acc, false, false);
            acc = __builtin_amdgcn_wmma_f32_16x16x32_bf16(false, alf[kc], false, bhb[cur], (short)0, acc, false, false);
            acc = __builtin_amdgcn_wmma_f32_16x16x32_bf16(false, ahf[kc], false, bhb[cur], (short)0, acc, false, false);
        }

        // Store with bias; NT hint keeps the 411MB output stream out of L2
        const int n = it * 32 + n0;
        const int p = n / W_DIM;
        const int w = n - p * W_DIM;
        const float bv = bias[w * P_DIM + p];
        const size_t base = ((size_t)(bb * P_DIM + p) * T_DIM + t0) * W_DIM + w;
        #pragma unroll
        for (int i = 0; i < 8; ++i) {
            __builtin_nontemporal_store(acc[i] + bv, &out[base + (size_t)i * W_DIM]);
        }

        BrH = nBrH;
        BrL = nBrL;
    }
}

extern "C" void kernel_launch(void* const* d_in, const int* in_sizes, int n_in,
                              void* d_out, int out_size, void* d_ws, size_t ws_size,
                              hipStream_t stream) {
    const float* q  = (const float*)d_in[0];   // [32,2048,256]
    const float* W  = (const float*)d_in[1];   // [49,32,256]
    const float* b  = (const float*)d_in[2];   // [49,32]
    float* out = (float*)d_out;

    unsigned short* WH = (unsigned short*)d_ws;            // 401408 shorts
    unsigned short* WL = WH + WELEMS;                      // +401408 shorts (~1.6MB total)

    // Pass 1: split weights once (100352 float4 / 256 threads = 392 blocks)
    split_weights<<<dim3(WELEMS / 4 / 256), dim3(256), 0, stream>>>(W, WH, WL);

    // Pass 2: 1024 blocks x 8 waves; each block owns 64 q-rows, sweeps all N
    decoder_head_gemm<<<dim3(65536 / M_TILE), dim3(256), 0, stream>>>(q, WH, WL, b, out);
}
